// DoubleCrossAttention_77257871720725
// MI455X (gfx1250) — matrix operand order repaired and checked by reading, not compiled, not run on hardware
//
#include <hip/hip_runtime.h>
#include <hip/hip_bf16.h>

// ---------------------------------------------------------------------------
// DoubleCrossAttention for MI455X (gfx1250), bf16 WMMA path.
// B=16, CL=2048, QL=512, H=1024.
// ---------------------------------------------------------------------------

#define B_   16
#define CL_  2048
#define QL_  512
#define H_   1024

typedef __attribute__((ext_vector_type(8)))  float   v8f;
typedef __attribute__((ext_vector_type(8)))  __bf16  v8bf;
typedef __attribute__((ext_vector_type(16))) __bf16  v16bf;

union Frag16 { v16bf f; v8bf h[2]; };

// ---------------------------------------------------------------------------
// Generic TN GEMM: D[m][n] = sum_k A[m][k] * Brow[n][k]
// A: M x K row-major (LDA), Brow: N x K row-major (LDB), both compile-time.
// Wave tile 64x64 (4x4 WMMA frags, 32 FLOP/byte), 4 waves -> 128x128 block.
// Optional outputs: f32 (ldoF), bf16 (ldoB), bf16 transposed (ldoT).
// ---------------------------------------------------------------------------
template<int LDA, int LDB>
__global__ __launch_bounds__(128) void gemm_tn_wmma(
    const __bf16* __restrict__ A,  long sA,
    const __bf16* __restrict__ Bm, long sB,
    float*  __restrict__ outF,  int ldoF, long sOF,
    __bf16* __restrict__ outB,  int ldoB, long sOB,
    __bf16* __restrict__ outT,  int ldoT, long sOT,
    int K)
{
    const int b    = blockIdx.z;
    const int wave = threadIdx.x >> 5;
    const int lane = threadIdx.x & 31;
    const int half = lane >> 4;       // 0 or 1
    const int lm   = lane & 15;

    const int m0 = blockIdx.x * 128 + (wave >> 1) * 64;
    const int n0 = blockIdx.y * 128 + (wave & 1) * 64;

    // A frag lane layout: row m0+16*fi+lm, K = 16*half + 0..15 contiguous
    const __bf16* ab = A  + (long)b * sA + (long)(m0 + lm) * LDA + 16 * half;
    // B frag lane layout: col n0+16*fj+lm, K = 16*half + 0..15 contiguous
    const __bf16* bb = Bm + (long)b * sB + (long)(n0 + lm) * LDB + 16 * half;

    v8f acc[4][4] = {};

    for (int k = 0; k < K; k += 32) {
        if (k + 64 < K) {  // prefetch two K-steps ahead (global_prefetch_b8)
            __builtin_prefetch(ab + k + 64, 0, 1);
            __builtin_prefetch(ab + 3 * 16 * LDA + k + 64, 0, 1);
            __builtin_prefetch(bb + k + 64, 0, 1);
            __builtin_prefetch(bb + 3 * 16 * LDB + k + 64, 0, 1);
        }
        Frag16 a[4], bf[4];
        #pragma unroll
        for (int fi = 0; fi < 4; ++fi) {
            a[fi].h[0] = *(const v8bf*)(ab + fi * 16 * LDA + k);
            a[fi].h[1] = *(const v8bf*)(ab + fi * 16 * LDA + k + 8);
        }
        #pragma unroll
        for (int fj = 0; fj < 4; ++fj)
            bf[fj].f = *(const v16bf*)(bb + fj * 16 * LDB + k);

        #pragma unroll
        for (int fi = 0; fi < 4; ++fi)
            #pragma unroll
            for (int fj = 0; fj < 4; ++fj)
                acc[fi][fj] = __builtin_amdgcn_wmma_f32_16x16x32_bf16(
                    false, a[fi].f, false, bf[fj].f, (short)0,
                    acc[fi][fj], false, false);
    }

    // C/D layout: VGPR r, lanes 0-15 -> M=r, lanes 16-31 -> M=r+8; N=lm.
    #pragma unroll
    for (int fi = 0; fi < 4; ++fi) {
        #pragma unroll
        for (int r = 0; r < 8; ++r) {
            const int grow = m0 + 16 * fi + r + 8 * half;
            #pragma unroll
            for (int fj = 0; fj < 4; ++fj) {
                const int gcol = n0 + 16 * fj + lm;
                const float v = acc[fi][fj][r];
                if (outF) outF[(long)b * sOF + (long)grow * ldoF + gcol] = v;
                if (outB) outB[(long)b * sOB + (long)grow * ldoB + gcol] = (__bf16)v;
                if (outT) outT[(long)b * sOT + (long)gcol * ldoT + grow] = (__bf16)v;
            }
        }
    }
}

// ---------------------------------------------------------------------------
// Convert f32 [rows][H] -> bf16 [rows][H] and bf16 transposed [H][rows];
// optionally copy f32 into output (concat slot 0), ld = ldcopy.
// LDS 32x32 tiled transpose, block (32,8), grid (H/32, rows/32, B).
// ---------------------------------------------------------------------------
__global__ __launch_bounds__(256) void convert_transpose(
    const float* __restrict__ src, int rows,
    float* __restrict__ outcopy, int ldcopy,
    __bf16* __restrict__ dst, __bf16* __restrict__ dstT)
{
    __shared__ float tile[32][33];
    const int b  = blockIdx.z;
    const int h0 = blockIdx.x * 32;
    const int r0 = blockIdx.y * 32;
    const int tx = threadIdx.x;
    const int ty = threadIdx.y;

    #pragma unroll
    for (int kk = 0; kk < 4; ++kk) {
        const int r = ty + 8 * kk;
        const float v = src[((long)b * rows + r0 + r) * H_ + h0 + tx];
        tile[r][tx] = v;
        dst[((long)b * rows + r0 + r) * H_ + h0 + tx] = (__bf16)v;
        if (outcopy)
            outcopy[((long)b * rows + r0 + r) * ldcopy + h0 + tx] = v;
    }
    __syncthreads();
    #pragma unroll
    for (int kk = 0; kk < 4; ++kk) {
        const int r = ty + 8 * kk;
        dstT[((long)b * H_ + h0 + r) * rows + r0 + tx] = (__bf16)tile[tx][r];
    }
}

// ---------------------------------------------------------------------------
// Row softmax over QL (axis=2), optional q_mask; writes bf16 probs.
// One 256-thread block per (row, batch); 2 elements per thread.
// ---------------------------------------------------------------------------
__global__ __launch_bounds__(256) void softmax_row(
    const float* __restrict__ S, const int* __restrict__ qmask,
    __bf16* __restrict__ P)
{
    __shared__ float red[256];
    const int i = blockIdx.x;
    const int b = blockIdx.y;
    const int t = threadIdx.x;

    const float* row = S + ((long)b * CL_ + i) * QL_;
    const int j0 = t, j1 = t + 256;
    const bool m0 = qmask ? (qmask[(long)b * QL_ + j0] != 0) : true;
    const bool m1 = qmask ? (qmask[(long)b * QL_ + j1] != 0) : true;
    const float x0 = m0 ? row[j0] : -1e30f;
    const float x1 = m1 ? row[j1] : -1e30f;

    red[t] = fmaxf(x0, x1);
    __syncthreads();
    for (int s2 = 128; s2 > 0; s2 >>= 1) {
        if (t < s2) red[t] = fmaxf(red[t], red[t + s2]);
        __syncthreads();
    }
    const float mx = red[0];
    __syncthreads();

    const float e0 = __expf(x0 - mx);
    const float e1 = __expf(x1 - mx);
    red[t] = e0 + e1;
    __syncthreads();
    for (int s2 = 128; s2 > 0; s2 >>= 1) {
        if (t < s2) red[t] += red[t + s2];
        __syncthreads();
    }
    const float inv = 1.0f / red[0];

    __bf16* prow = P + ((long)b * CL_ + i) * QL_;
    prow[j0] = (__bf16)(e0 * inv);
    prow[j1] = (__bf16)(e1 * inv);
}

// ---------------------------------------------------------------------------
// Column softmax over CL (axis=1) with c_mask; writes TRANSPOSED bf16
// probs: betaT[b][j][i]. One thread per column j, online max/sum, then
// packed 16-element (32B) row-slice writes.
// Grid (QL/256, B), block 256.
// ---------------------------------------------------------------------------
__global__ __launch_bounds__(256) void softmax_col_T(
    const float* __restrict__ S, const int* __restrict__ cmask,
    __bf16* __restrict__ betaT)
{
    const int b = blockIdx.y;
    const int j = blockIdx.x * 256 + threadIdx.x;
    const float* Sb = S + (long)b * CL_ * QL_;
    const int* cm = cmask + (long)b * CL_;

    float m = -3.0e38f, l = 0.0f;
    for (int i = 0; i < CL_; ++i) {
        const float x = cm[i] ? Sb[(long)i * QL_ + j] : -1e30f;
        if (x > m) { l *= __expf(m - x); m = x; }
        l += __expf(x - m);
    }
    const float inv = 1.0f / l;

    __bf16* drow = betaT + ((long)b * QL_ + j) * CL_;
    for (int i0 = 0; i0 < CL_; i0 += 16) {
        v8bf p0, p1;
        #pragma unroll
        for (int ii = 0; ii < 16; ++ii) {
            const int i = i0 + ii;
            const float x = cm[i] ? Sb[(long)i * QL_ + j] : -1e30f;
            const float v = __expf(x - m) * inv;
            if (ii < 8) p0[ii] = (__bf16)v; else p1[ii - 8] = (__bf16)v;
        }
        *(v8bf*)(drow + i0)     = p0;
        *(v8bf*)(drow + i0 + 8) = p1;
    }
}

// ---------------------------------------------------------------------------
// Host-side orchestration. Workspace layout (bytes):
// ---------------------------------------------------------------------------
static constexpr size_t OFF_C16  = 0;                                // [B][CL][H] bf16
static constexpr size_t OFF_CT16 = OFF_C16  + (size_t)B_*CL_*H_*2;   // [B][H][CL] bf16
static constexpr size_t OFF_Q16  = OFF_CT16 + (size_t)B_*CL_*H_*2;   // [B][QL][H] bf16
static constexpr size_t OFF_QT16 = OFF_Q16  + (size_t)B_*QL_*H_*2;   // [B][H][QL] bf16
static constexpr size_t OFF_S    = OFF_QT16 + (size_t)B_*QL_*H_*2;   // [B][CL][QL] f32 (s, then R)
static constexpr size_t OFF_P    = OFF_S    + (size_t)B_*CL_*QL_*4;  // [B][CL][QL] bf16 (alpha, then gamma)
static constexpr size_t OFF_BT   = OFF_P    + (size_t)B_*CL_*QL_*2;  // [B][QL][CL] bf16 (betaT)
static constexpr size_t OFF_A16  = OFF_BT   + (size_t)B_*CL_*QL_*2;  // [B][CL][H] bf16
static constexpr size_t OFF_BM   = OFF_A16  + (size_t)B_*CL_*H_*2;   // [B][QL][H] bf16 (Bmat)
static constexpr size_t OFF_BMT  = OFF_BM   + (size_t)B_*QL_*H_*2;   // [B][H][QL] bf16 (BmatT)

extern "C" void kernel_launch(void* const* d_in, const int* in_sizes, int n_in,
                              void* d_out, int out_size, void* d_ws, size_t ws_size,
                              hipStream_t stream) {
    (void)in_sizes; (void)n_in; (void)out_size; (void)ws_size;
    const float* c      = (const float*)d_in[0];
    const float* q      = (const float*)d_in[1];
    const int*   c_mask = (const int*)d_in[2];
    const int*   q_mask = (const int*)d_in[3];
    float* out = (float*)d_out;
    char*  ws  = (char*)d_ws;

    __bf16* c16  = (__bf16*)(ws + OFF_C16);
    __bf16* cT16 = (__bf16*)(ws + OFF_CT16);
    __bf16* q16  = (__bf16*)(ws + OFF_Q16);
    __bf16* qT16 = (__bf16*)(ws + OFF_QT16);
    float*  S    = (float*) (ws + OFF_S);
    __bf16* P    = (__bf16*)(ws + OFF_P);
    __bf16* bT   = (__bf16*)(ws + OFF_BT);
    __bf16* A16  = (__bf16*)(ws + OFF_A16);
    __bf16* Bm16 = (__bf16*)(ws + OFF_BM);
    __bf16* BmT  = (__bf16*)(ws + OFF_BMT);

    const dim3 blkT(32, 8);
    const dim3 blkG(128);
    const dim3 blkS(256);

    // 1) c -> c16, cT16, and copy into out[..., 0:H]
    convert_transpose<<<dim3(H_/32, CL_/32, B_), blkT, 0, stream>>>(
        c, CL_, out, 3 * H_, c16, cT16);
    // 2) q -> q16, qT16
    convert_transpose<<<dim3(H_/32, QL_/32, B_), blkT, 0, stream>>>(
        q, QL_, (float*)nullptr, 0, q16, qT16);

    // 3) s = c . q^T   [CL x QL], K=H
    gemm_tn_wmma<H_, H_><<<dim3(CL_/128, QL_/128, B_), blkG, 0, stream>>>(
        c16, (long)CL_*H_,  q16, (long)QL_*H_,
        S, QL_, (long)CL_*QL_,
        (__bf16*)nullptr, 0, 0, (__bf16*)nullptr, 0, 0, H_);

    // 4) beta = softmax_col(s, c_mask) -> betaT [QL][CL]
    softmax_col_T<<<dim3(QL_/256, B_), blkS, 0, stream>>>(S, c_mask, bT);
    // 5) alpha = softmax_row(s, q_mask) -> P [CL][QL]
    softmax_row<<<dim3(CL_, B_), blkS, 0, stream>>>(S, q_mask, P);

    // 6) Bmat = beta^T . c  [QL x H], K=CL; store Bmat and Bmat^T (bf16)
    gemm_tn_wmma<CL_, CL_><<<dim3(QL_/128, H_/128, B_), blkG, 0, stream>>>(
        bT, (long)QL_*CL_,  cT16, (long)H_*CL_,
        (float*)nullptr, 0, 0,
        Bm16, H_, (long)QL_*H_,  BmT, QL_, (long)H_*QL_, CL_);

    // 7) A = alpha . q  [CL x H], K=QL; f32 -> out[..., H:2H], bf16 -> A16
    gemm_tn_wmma<QL_, QL_><<<dim3(CL_/128, H_/128, B_), blkG, 0, stream>>>(
        P, (long)CL_*QL_,  qT16, (long)H_*QL_,
        out + H_, 3 * H_, (long)CL_*3*H_,
        A16, H_, (long)CL_*H_,  (__bf16*)nullptr, 0, 0, QL_);

    // 8) R = A . Bmat^T  [CL x QL], K=H -> reuse S
    gemm_tn_wmma<H_, H_><<<dim3(CL_/128, QL_/128, B_), blkG, 0, stream>>>(
        A16, (long)CL_*H_,  Bm16, (long)QL_*H_,
        S, QL_, (long)CL_*QL_,
        (__bf16*)nullptr, 0, 0, (__bf16*)nullptr, 0, 0, H_);

    // 9) gamma = softmax_row(R) (unmasked) -> P
    softmax_row<<<dim3(CL_, B_), blkS, 0, stream>>>(S, (const int*)nullptr, P);

    // 10) D = gamma . Bmat  [CL x H], K=QL; f32 -> out[..., 2H:3H]
    gemm_tn_wmma<QL_, QL_><<<dim3(CL_/128, H_/128, B_), blkG, 0, stream>>>(
        P, (long)CL_*QL_,  BmT, (long)H_*QL_,
        out + 2 * H_, 3 * H_, (long)CL_*3*H_,
        (__bf16*)nullptr, 0, 0, (__bf16*)nullptr, 0, 0, QL_);
}
